// PNAPCSAFT_19035295055923
// MI455X (gfx1250) — compile-verified, matrix operand order checked by reading, not run on hardware
//
#include <hip/hip_runtime.h>
#include <hip/hip_bf16.h>

// ---------------------------------------------------------------------------
// PNA GNN forward for MI455X (gfx1250): all GEMMs via v_wmma_f32_16x16x32_bf16
// (bf16 inputs, fp32 accumulate). Weights repacked into WMMA-fragment order so
// every B fragment is two coalesced global_load_b128. Segment stats scattered
// into one interleaved 16B record per element (sum|sumsq|minenc|maxenc) so all
// four atomics share one address. Edge-feature staging uses async global->LDS
// loads (ASYNCcnt) when the toolchain exposes the builtin.
// ---------------------------------------------------------------------------

#define N_NODES 16384
#define N_EDGES 131072
#define N_GRAPHS 512
#define HIDDIM 128
#define AVG_LOG_DEG 2.19722457733622f   // log(9)

#if __has_builtin(__builtin_amdgcn_global_load_async_to_lds_b128) && \
    __has_builtin(__builtin_amdgcn_s_wait_asynccnt)
#define HAVE_ASYNC_LDS 1
#else
#define HAVE_ASYNC_LDS 0
#endif

typedef __attribute__((ext_vector_type(16))) __bf16 v16bf;
typedef __attribute__((ext_vector_type(8)))  float  v8f;
typedef __attribute__((__vector_size__(16))) int    vs4i;   // builtin's b128 type

// ---------------- helpers ----------------
__device__ __forceinline__ __bf16 bfu(unsigned short u) {
  union { unsigned short u; __bf16 b; } x; x.u = u; return x.b;
}
__device__ __forceinline__ unsigned short f2bu(float f) {   // fp32 -> bf16 RNE
  unsigned u = __float_as_uint(f);
  return (unsigned short)((u + 0x7FFFu + ((u >> 16) & 1u)) >> 16);
}
__device__ __forceinline__ unsigned ordenc(float f) {       // order-preserving
  unsigned u = __float_as_uint(f);
  return (u & 0x80000000u) ? ~u : (u | 0x80000000u);
}
__device__ __forceinline__ float orddec(unsigned e) {
  unsigned u = (e & 0x80000000u) ? (e & 0x7FFFFFFFu) : ~e;
  return __uint_as_float(u);
}

__device__ __forceinline__ v8f wmma_bf16(v16bf a, v16bf b, v8f c) {
  return __builtin_amdgcn_wmma_f32_16x16x32_bf16(false, a, false, b,
                                                 (short)0, c, false, false);
}

// WMMA 16-bit fragment K-offset pattern: lane<16 -> K {0..7,16..23}+ks,
// lane>=16 -> K {8..15,24..31}+ks (rows/cols = lane&15).

// swizzled offset of logical (k, c) inside a [K][N] B matrix packed in
// fragment order: per (kb, nb) block, lane-major, 16 contiguous halfs per lane.
__device__ __forceinline__ size_t swz(int K, int k, int c) {
  int kb = k >> 5, ko = k & 31;
  int nb = c >> 4, cn = c & 15;
  int sel = (ko >> 3) & 1;
  int j = (ko & 7) | ((ko >> 4) << 3);
  int lane = (sel << 4) | cn;
  return ((((size_t)nb * (K >> 5) + kb) * 32) + lane) * 16 + j;
}

// B fragment from swizzled matrix: 2 x 16B contiguous loads per lane.
__device__ __forceinline__ v16bf frag_b_sw(const unsigned short* __restrict__ mat,
                                           int KD32, int nb, int kb, int lane,
                                           bool prefetch_next) {
  const unsigned short* p = mat + (((size_t)nb * KD32 + kb) * 32 + lane) * 16;
  if (prefetch_next) __builtin_prefetch(p + 32 * 16, 0, 1);
  union { uint4 q; unsigned short u[8]; } lo, hi;
  lo.q = *(const uint4*)p;
  hi.q = *(const uint4*)(p + 8);
  v16bf b;
#pragma unroll
  for (int j = 0; j < 8; ++j) { b[j] = bfu(lo.u[j]); b[8 + j] = bfu(hi.u[j]); }
  return b;
}

// A fragment (16xK row-major bf16 tile in LDS)
template <int STRIDE>
__device__ __forceinline__ v16bf frag_a_lds(const unsigned short* t0, int ks, int lane) {
  const int row = lane & 15;
  const int sel = (lane >> 4) << 3;
  const unsigned short* p = t0 + row * STRIDE + ks + sel;
  v16bf a;
#pragma unroll
  for (int j = 0; j < 8; ++j) a[j] = bfu(p[j]);
#pragma unroll
  for (int j = 0; j < 8; ++j) a[8 + j] = bfu(p[16 + j]);
  return a;
}

// ---------------- small utility kernels ----------------
__global__ void zero_f32_k(float* p, long n) {
  long i = (long)blockIdx.x * blockDim.x + threadIdx.x;
  if (i < n) p[i] = 0.f;
}
// accumulator record init: {sum=0, sumsq=0, minenc=MAX, maxenc=0}
__global__ void fill_acc_k(uint4* p, long n) {
  long i = (long)blockIdx.x * blockDim.x + threadIdx.x;
  if (i < n) p[i] = make_uint4(0u, 0u, 0xFFFFFFFFu, 0u);
}
// x0: [N,9] fp32 -> [N,32] bf16 zero-padded
__global__ void pack_x0_k(const float* __restrict__ x, unsigned short* __restrict__ o) {
  int i = blockIdx.x * blockDim.x + threadIdx.x;
  if (i >= N_NODES * 32) return;
  int n = i >> 5, k = i & 31;
  o[i] = (k < 9) ? f2bu(x[n * 9 + k]) : (unsigned short)0;
}
// Wpre [T,3F,F] -> swizzled padded bf16 [T][KPRE x Fp]
__global__ void pack_wpre_k(const float* __restrict__ W, unsigned short* __restrict__ o,
                            int F, int Fp, int KPRE) {
  int idx = blockIdx.x * blockDim.x + threadIdx.x;
  int total = 4 * KPRE * Fp;
  if (idx >= total) return;
  int t = idx / (KPRE * Fp);
  int r = (idx / Fp) % KPRE;
  int c = idx % Fp;
  float v = 0.f;
  if (r < 3 * F && c < F) v = W[((size_t)t * 3 * F + r) * F + c];
  o[(size_t)t * KPRE * Fp + swz(KPRE, r, c)] = f2bu(v);
}
// Wpost [T,13F,32] -> swizzled padded bf16 [T][KPOST x 32];
// padded K = [x(KX) | agg(4Fp) | agg*s1(4Fp) | agg*s2(4Fp)]
__global__ void pack_wpost_k(const float* __restrict__ W, unsigned short* __restrict__ o,
                             int F, int Fp, int KX, int KPOST) {
  int idx = blockIdx.x * blockDim.x + threadIdx.x;
  int total = 4 * KPOST * 32;
  if (idx >= total) return;
  int t = idx / (KPOST * 32);
  int r = (idx / 32) % KPOST;
  int c = idx % 32;
  float v = 0.f;
  if (r < KX) {
    if (r < F) v = W[((size_t)t * 13 * F + r) * 32 + c];
  } else {
    int k2 = r - KX;
    int sg = k2 / (4 * Fp);
    int rem = k2 % (4 * Fp);
    int stat = rem / Fp, f = rem % Fp;
    if (f < F) {
      int pos = F + sg * 4 * F + stat * F + f;
      v = W[((size_t)t * 13 * F + pos) * 32 + c];
    }
  }
  o[(size_t)t * KPOST * 32 + swz(KPOST, r, c)] = f2bu(v);
}
// Wlin [128,128] -> swizzled bf16
__global__ void pack_wlin_k(const float* __restrict__ W, unsigned short* __restrict__ o) {
  int idx = blockIdx.x * blockDim.x + threadIdx.x;
  if (idx >= 128 * 128) return;
  int k = idx >> 7, c = idx & 127;
  o[swz(128, k, c)] = f2bu(W[idx]);
}

// ---------------- edge kernel: pre-MLP GEMM + scatter stats ----------------
template <int F, int Fp, int KPRE, int KX>
__global__ __launch_bounds__(256) void pna_edge_kernel(
    const unsigned short* __restrict__ xb,
    const int* __restrict__ src, const int* __restrict__ dst,
    const float* __restrict__ ea, const float* __restrict__ We,
    const float* __restrict__ be, const unsigned short* __restrict__ wpre,
    const float* __restrict__ bpre, float* __restrict__ acc,
    float* __restrict__ cnt) {
  __shared__ unsigned short hsm[16 * KPRE];
  __shared__ int sidx[16], didx[16];
  const int tid = threadIdx.x;
  const int ebase = blockIdx.x * 16;
  if (tid < 16) { sidx[tid] = src[ebase + tid]; didx[tid] = dst[ebase + tid]; }
  __syncthreads();

  // build h = [x_dst | x_src | e] tile in LDS (bf16)
  if constexpr (F % 8 == 0) {
#if HAVE_ASYNC_LDS
    {  // x_dst / x_src sections: async 16B copies global -> LDS (ASYNCcnt)
      const int XR = (2 * F) / 8;
      for (int idx = tid; idx < 16 * XR; idx += 256) {
        int r = idx / XR, k0 = (idx - r * XR) * 8;
        const unsigned short* gp = (k0 < F)
            ? xb + (size_t)didx[r] * KX + k0
            : xb + (size_t)sidx[r] * KX + (k0 - F);
        unsigned short* lp = hsm + r * KPRE + k0;
        __builtin_amdgcn_global_load_async_to_lds_b128(
            (__attribute__((address_space(1))) vs4i*)(unsigned long long)(__SIZE_TYPE__)gp,
            (__attribute__((address_space(3))) vs4i*)(unsigned)(__SIZE_TYPE__)lp,
            0, 0);
      }
      // e section (computed) + zero padding
      const int RUNS = KPRE / 8;
      const int ER = RUNS - XR;
      for (int idx = tid; idx < 16 * ER; idx += 256) {
        int r = idx / ER, k0 = 2 * F + (idx - r * ER) * 8;
        union { uint4 q; unsigned short u[8]; } x;
        if (k0 < 3 * F) {
          size_t e = (size_t)(ebase + r) * 3;
          float a0 = ea[e], a1 = ea[e + 1], a2 = ea[e + 2];
#pragma unroll
          for (int j = 0; j < 8; ++j) {
            int f = k0 - 2 * F + j;
            x.u[j] = f2bu(a0 * We[f] + a1 * We[F + f] + a2 * We[2 * F + f] + be[f]);
          }
        } else {
#pragma unroll
          for (int j = 0; j < 8; ++j) x.u[j] = 0;
        }
        *(uint4*)(hsm + r * KPRE + k0) = x.q;
      }
      __builtin_amdgcn_s_wait_asynccnt(0);
    }
#else
    const int RUNS = KPRE / 8;
    for (int idx = tid; idx < 16 * RUNS; idx += 256) {
      int r = idx / RUNS, k0 = (idx - r * RUNS) * 8;
      union { uint4 q; unsigned short u[8]; } x;
      if (k0 < F) {
        x.q = *(const uint4*)(xb + (size_t)didx[r] * KX + k0);
      } else if (k0 < 2 * F) {
        x.q = *(const uint4*)(xb + (size_t)sidx[r] * KX + (k0 - F));
      } else if (k0 < 3 * F) {
        size_t e = (size_t)(ebase + r) * 3;
        float a0 = ea[e], a1 = ea[e + 1], a2 = ea[e + 2];
#pragma unroll
        for (int j = 0; j < 8; ++j) {
          int f = k0 - 2 * F + j;
          x.u[j] = f2bu(a0 * We[f] + a1 * We[F + f] + a2 * We[2 * F + f] + be[f]);
        }
      } else {
#pragma unroll
        for (int j = 0; j < 8; ++j) x.u[j] = 0;
      }
      *(uint4*)(hsm + r * KPRE + k0) = x.q;
    }
#endif
  } else {
    for (int idx = tid; idx < 16 * KPRE; idx += 256) {
      int r = idx / KPRE, k = idx - r * KPRE;
      unsigned short u = 0;
      if (k < F) u = xb[(size_t)didx[r] * KX + k];
      else if (k < 2 * F) u = xb[(size_t)sidx[r] * KX + (k - F)];
      else if (k < 3 * F) {
        int f = k - 2 * F;
        size_t e = (size_t)(ebase + r) * 3;
        float v = ea[e] * We[f] + ea[e + 1] * We[F + f] + ea[e + 2] * We[2 * F + f] + be[f];
        u = f2bu(v);
      }
      hsm[r * KPRE + k] = u;
    }
  }
  __syncthreads();

  const int w = tid >> 5, lane = tid & 31;
  const int NTOT = (4 * Fp) / 16;  // output col tiles over T*Fp
  for (int nt = w; nt < NTOT; nt += 8) {
    const int t = (nt * 16) / Fp;
    const int nb = ((nt * 16) % Fp) >> 4;
    const unsigned short* wp = wpre + (size_t)t * KPRE * Fp;
    v8f c;
#pragma unroll
    for (int j = 0; j < 8; ++j) c[j] = 0.f;
#pragma unroll 4
    for (int ks = 0; ks < KPRE; ks += 32) {
      v16bf a = frag_a_lds<KPRE>(hsm, ks, lane);
      v16bf b = frag_b_sw(wp, KPRE / 32, nb, ks >> 5, lane, ks + 32 < KPRE);
      c = wmma_bf16(a, b, c);
    }
    const int colw = (nb << 4) + (lane & 15);
    const int gcol = t * Fp + colw;
    const float bias = (colw < F) ? bpre[t * F + colw] : 0.f;
#pragma unroll
    for (int r = 0; r < 8; ++r) {
      int m = r + ((lane >> 4) << 3);
      float v = c[r] + bias;
      // one interleaved 16B record per element: all 4 atomics share one address
      float* a4 = acc + (((size_t)didx[m] * (4 * Fp) + gcol) << 2);
      atomicAdd(a4 + 0, v);
      atomicAdd(a4 + 1, v * v);
      unsigned enc = ordenc(v);
      atomicMin((unsigned*)a4 + 2, enc);
      atomicMax((unsigned*)a4 + 3, enc);
    }
  }
  if (tid < 16) atomicAdd(cnt + didx[tid], 1.0f);
}

// ---------------- per-node degree scalers ----------------
__global__ void scalers_k(const float* __restrict__ cnt, float* __restrict__ s1,
                          float* __restrict__ s2) {
  int n = blockIdx.x * blockDim.x + threadIdx.x;
  if (n >= N_NODES) return;
  float d = fmaxf(cnt[n], 1.f);
  float logd = logf(d + 1.f);
  s1[n] = logd / AVG_LOG_DEG;
  s2[n] = AVG_LOG_DEG / logd;
}

// ---------------- stats: accumulators -> bf16 scaled feature block ----------
// xpost[(n*4+t)*12Fp + ...] = [mean|mn|mx|std | x s1 | x s2] (each Fp wide)
__global__ void stats_k(const uint4* __restrict__ acc, const float* __restrict__ cnt,
                        const float* __restrict__ s1, const float* __restrict__ s2,
                        unsigned short* __restrict__ xpost, int Fp) {
  const int TFp = 4 * Fp;
  long idx = (long)blockIdx.x * blockDim.x + threadIdx.x;
  if (idx >= (long)N_NODES * TFp) return;
  int n = (int)(idx / TFp);
  int c = (int)(idx - (long)n * TFp);
  uint4 a = acc[idx];
  float cn = cnt[n];
  float denom = fmaxf(cn, 1.f);
  float mean = __uint_as_float(a.x) / denom;
  float ms = __uint_as_float(a.y) / denom;
  float var = ms - mean * mean;
  if (var < 0.f) var = 0.f;
  float sd = sqrtf(var + 1e-5f);
  float mn = 0.f, mx = 0.f;
  if (cn > 0.f) { mn = orddec(a.z); mx = orddec(a.w); }
  float s1v = s1[n], s2v = s2[n];
  int t = c / Fp, f = c - t * Fp;
  size_t b = ((size_t)n * 4 + t) * (size_t)(12 * Fp) + f;
  xpost[b + 0 * Fp] = f2bu(mean);
  xpost[b + 1 * Fp] = f2bu(mn);
  xpost[b + 2 * Fp] = f2bu(mx);
  xpost[b + 3 * Fp] = f2bu(sd);
  xpost[b + 4 * Fp] = f2bu(mean * s1v);
  xpost[b + 5 * Fp] = f2bu(mn * s1v);
  xpost[b + 6 * Fp] = f2bu(mx * s1v);
  xpost[b + 7 * Fp] = f2bu(sd * s1v);
  xpost[b + 8 * Fp] = f2bu(mean * s2v);
  xpost[b + 9 * Fp] = f2bu(mn * s2v);
  xpost[b + 10 * Fp] = f2bu(mx * s2v);
  xpost[b + 11 * Fp] = f2bu(sd * s2v);
}

// ---------------- node kernel: post-MLP WMMA -> Wlin WMMA -> LN+ReLU --------
template <int Fp, int KX, int KPOST>
__global__ __launch_bounds__(256) void pna_node_kernel(
    const unsigned short* __restrict__ xb, const unsigned short* __restrict__ xpost,
    const unsigned short* __restrict__ wpost, const float* __restrict__ bpost,
    const unsigned short* __restrict__ wlin, const float* __restrict__ blin,
    const float* __restrict__ lng, const float* __restrict__ lnb,
    unsigned short* __restrict__ xb_out, float* __restrict__ xf_out) {
  __shared__ unsigned short hsn[16 * HIDDIM];  // post-MLP output (bf16)
  __shared__ float y[16 * HIDDIM];             // Wlin output (fp32)
  __shared__ float mu[16], rs[16];
  const int tid = threadIdx.x;
  const int w = tid >> 5, lane = tid & 31;
  const int base = blockIdx.x * 16;

  // Phase A: wave w -> cols [w*16, w*16+16) of the T*32=128 post-MLP output
  {
    const int t = w >> 1;            // tower (32 cols each)
    const int nb = w & 1;
    const int node = base + (lane & 15);
    const unsigned short* rowx = xb + (size_t)node * KX;
    const unsigned short* rowp = xpost + ((size_t)node * 4 + t) * (size_t)(12 * Fp);
    const unsigned short* wp = wpost + (size_t)t * KPOST * 32;
    const int sel = (lane >> 4) << 3;
    v8f c;
#pragma unroll
    for (int j = 0; j < 8; ++j) c[j] = 0.f;
#pragma unroll 4
    for (int ks = 0; ks < KPOST; ks += 32) {
      v16bf a;
      {  // 8-runs at ks+sel and ks+16+sel never straddle the KX boundary
        int k0 = ks + sel;
        const unsigned short* p0 = (k0 < KX) ? (rowx + k0) : (rowp + (k0 - KX));
        union { uint4 q; unsigned short u[8]; } x0;
        x0.q = *(const uint4*)p0;
        int k1 = ks + 16 + sel;
        const unsigned short* p1 = (k1 < KX) ? (rowx + k1) : (rowp + (k1 - KX));
        union { uint4 q; unsigned short u[8]; } x1;
        x1.q = *(const uint4*)p1;
#pragma unroll
        for (int j = 0; j < 8; ++j) { a[j] = bfu(x0.u[j]); a[8 + j] = bfu(x1.u[j]); }
      }
      v16bf b = frag_b_sw(wp, KPOST / 32, nb, ks >> 5, lane, ks + 32 < KPOST);
      c = wmma_bf16(a, b, c);
    }
    const int colw = (nb << 4) + (lane & 15);
    const float bias = bpost[t * 32 + colw];
#pragma unroll
    for (int r = 0; r < 8; ++r) {
      int m = r + ((lane >> 4) << 3);
      hsn[m * HIDDIM + t * 32 + colw] = f2bu(c[r] + bias);
    }
  }
  __syncthreads();

  // Phase B: y = hs @ Wlin + blin (K=128, wave w -> 16 output cols)
  {
    v8f c;
#pragma unroll
    for (int j = 0; j < 8; ++j) c[j] = 0.f;
#pragma unroll
    for (int ks = 0; ks < HIDDIM; ks += 32) {
      v16bf a = frag_a_lds<HIDDIM>(hsn, ks, lane);
      v16bf b = frag_b_sw(wlin, 4, w, ks >> 5, lane, false);
      c = wmma_bf16(a, b, c);
    }
    const int col = w * 16 + (lane & 15);
    const float bias = blin[col];
#pragma unroll
    for (int r = 0; r < 8; ++r) {
      int m = r + ((lane >> 4) << 3);
      y[m * HIDDIM + col] = c[r] + bias;
    }
  }
  __syncthreads();

  // Phase C: LayerNorm + ReLU
  if (tid < 16) {
    float s = 0.f;
    for (int c = 0; c < HIDDIM; ++c) s += y[tid * HIDDIM + c];
    float m = s / HIDDIM;
    float v = 0.f;
    for (int c = 0; c < HIDDIM; ++c) { float d = y[tid * HIDDIM + c] - m; v += d * d; }
    mu[tid] = m;
    rs[tid] = rsqrtf(v / HIDDIM + 1e-5f);
  }
  __syncthreads();
  for (int idx = tid; idx < 16 * HIDDIM; idx += 256) {
    int m = idx >> 7, cc = idx & 127;
    float v = (y[idx] - mu[m]) * rs[m] * lng[cc] + lnb[cc];
    v = fmaxf(v, 0.f);
    int node = base + m;
    xf_out[(size_t)node * HIDDIM + cc] = v;
    xb_out[(size_t)node * HIDDIM + cc] = f2bu(v);
  }
}

// ---------------- pooling + head ----------------
__global__ void pool_k(const float* __restrict__ xf, const int* __restrict__ batch,
                       float* __restrict__ g) {
  long idx = (long)blockIdx.x * blockDim.x + threadIdx.x;
  if (idx >= (long)N_NODES * HIDDIM) return;
  int n = (int)(idx >> 7), c = (int)(idx & 127);
  atomicAdd(g + (size_t)batch[n] * HIDDIM + c, xf[idx]);
}

// mode: 0 = none, 1 = relu, 2 = abs
__global__ void gemm_head_k(const float* __restrict__ A, const float* __restrict__ B,
                            const float* __restrict__ bias, float* __restrict__ C,
                            int M, int K, int Nc, int mode) {
  int idx = blockIdx.x * blockDim.x + threadIdx.x;
  if (idx >= M * Nc) return;
  int row = idx / Nc, col = idx - row * Nc;
  float acc = bias[col];
  for (int k = 0; k < K; ++k) acc += A[(size_t)row * K + k] * B[(size_t)k * Nc + col];
  if (mode == 1) acc = fmaxf(acc, 0.f);
  else if (mode == 2) acc = fabsf(acc);
  C[idx] = acc;
}

__global__ void ln_relu_head_k(float* __restrict__ X, const float* __restrict__ g,
                               const float* __restrict__ b, int M) {
  int row = blockIdx.x * blockDim.x + threadIdx.x;
  if (row >= M) return;
  float* p = X + (size_t)row * HIDDIM;
  float s = 0.f;
  for (int c = 0; c < HIDDIM; ++c) s += p[c];
  float mu = s / HIDDIM;
  float v = 0.f;
  for (int c = 0; c < HIDDIM; ++c) { float d = p[c] - mu; v += d * d; }
  float rs = rsqrtf(v / HIDDIM + 1e-5f);
  for (int c = 0; c < HIDDIM; ++c)
    p[c] = fmaxf((p[c] - mu) * rs * g[c] + b[c], 0.f);
}

// ---------------------------------------------------------------------------
extern "C" void kernel_launch(void* const* d_in, const int* in_sizes, int n_in,
                              void* d_out, int out_size, void* d_ws, size_t ws_size,
                              hipStream_t stream) {
  (void)in_sizes; (void)n_in; (void)out_size; (void)ws_size;
  // Input layout (top-level insertion order; params via jax-style sorted keys):
  // 0:x 1:edge_index 2:edge_attr 3:batch
  // convs[l] @ 4+10l: We,Wlin,Wpost,Wpre,be,blin,bpost,bpre,ln_b,ln_g
  // mlp @ 34: W1,W2,b1,b2,be1,be2,g1,g2 ; out @ 42: W1,W2,W3,b1,b2,b3
  const float* x0   = (const float*)d_in[0];
  const int*   eidx = (const int*)d_in[1];
  const float* ea   = (const float*)d_in[2];
  const int*   batch = (const int*)d_in[3];
  const int* src = eidx;
  const int* dst = eidx + N_EDGES;

  // workspace carve-up
  char* wsp = (char*)d_ws;
  size_t off = 0;
  auto take = [&](size_t bytes) -> void* {
    void* p = wsp + off;
    off += (bytes + 255) & ~(size_t)255;
    return p;
  };
  unsigned short* xbA   = (unsigned short*)take((size_t)N_NODES * 128 * 2);
  unsigned short* xbB   = (unsigned short*)take((size_t)N_NODES * 128 * 2);
  float* xf             = (float*)take((size_t)N_NODES * 128 * 4);
  float* acc            = (float*)take((size_t)N_NODES * 512 * 16); // interleaved records
  float* cnt            = (float*)take((size_t)N_NODES * 4);
  float* s1             = (float*)take((size_t)N_NODES * 4);
  float* s2             = (float*)take((size_t)N_NODES * 4);
  unsigned short* xpost = (unsigned short*)take((size_t)N_NODES * 4 * 12 * 128 * 2);
  unsigned short* wpreP = (unsigned short*)take((size_t)4 * 384 * 128 * 2);
  unsigned short* wpostP= (unsigned short*)take((size_t)4 * 1664 * 32 * 2);
  unsigned short* wlinP = (unsigned short*)take((size_t)128 * 128 * 2);
  float* gpool          = (float*)take((size_t)N_GRAPHS * 128 * 4);
  float* h1             = (float*)take((size_t)N_GRAPHS * 128 * 4);
  float* h2             = (float*)take((size_t)N_GRAPHS * 128 * 4);
  float* o1             = (float*)take((size_t)N_GRAPHS * 64 * 4);
  float* o2             = (float*)take((size_t)N_GRAPHS * 32 * 4);

  // x -> bf16 padded [N][32]
  pack_x0_k<<<(N_NODES * 32 + 255) / 256, 256, 0, stream>>>(x0, xbA);

  unsigned short* xb_in = xbA;
  unsigned short* xb_out = xbB;

  for (int l = 0; l < 3; ++l) {
    const float* We   = (const float*)d_in[4 + l * 10 + 0];
    const float* Wlin = (const float*)d_in[4 + l * 10 + 1];
    const float* Wpost= (const float*)d_in[4 + l * 10 + 2];
    const float* Wpre = (const float*)d_in[4 + l * 10 + 3];
    const float* be   = (const float*)d_in[4 + l * 10 + 4];
    const float* blin = (const float*)d_in[4 + l * 10 + 5];
    const float* bpost= (const float*)d_in[4 + l * 10 + 6];
    const float* bpre = (const float*)d_in[4 + l * 10 + 7];
    const float* ln_b = (const float*)d_in[4 + l * 10 + 8];
    const float* ln_g = (const float*)d_in[4 + l * 10 + 9];

    const int F     = (l == 0) ? 9 : 128;
    const int Fp    = (l == 0) ? 16 : 128;
    const int KPRE  = (l == 0) ? 32 : 384;
    const int KX    = (l == 0) ? 32 : 128;
    const int KPOST = (l == 0) ? 224 : 1664;
    const long accN = (long)N_NODES * 4 * Fp;

    // pack weights to swizzled bf16 WMMA layouts
    pack_wpre_k<<<(4 * KPRE * Fp + 255) / 256, 256, 0, stream>>>(Wpre, wpreP, F, Fp, KPRE);
    pack_wpost_k<<<(4 * KPOST * 32 + 255) / 256, 256, 0, stream>>>(Wpost, wpostP, F, Fp, KX, KPOST);
    pack_wlin_k<<<(128 * 128 + 255) / 256, 256, 0, stream>>>(Wlin, wlinP);

    // init accumulator records + counts
    fill_acc_k<<<(int)((accN + 255) / 256), 256, 0, stream>>>((uint4*)acc, accN);
    zero_f32_k<<<(N_NODES + 255) / 256, 256, 0, stream>>>(cnt, N_NODES);

    // edge pre-MLP + scatter stats
    if (l == 0) {
      pna_edge_kernel<9, 16, 32, 32><<<N_EDGES / 16, 256, 0, stream>>>(
          xb_in, src, dst, ea, We, be, wpreP, bpre, acc, cnt);
    } else {
      pna_edge_kernel<128, 128, 384, 128><<<N_EDGES / 16, 256, 0, stream>>>(
          xb_in, src, dst, ea, We, be, wpreP, bpre, acc, cnt);
    }

    scalers_k<<<(N_NODES + 255) / 256, 256, 0, stream>>>(cnt, s1, s2);
    stats_k<<<(int)((accN + 255) / 256), 256, 0, stream>>>((const uint4*)acc, cnt,
                                                           s1, s2, xpost, Fp);

    // node post-MLP + Wlin + LN + ReLU
    if (l == 0) {
      pna_node_kernel<16, 32, 224><<<N_NODES / 16, 256, 0, stream>>>(
          xb_in, xpost, wpostP, bpost, wlinP, blin, ln_g, ln_b, xb_out, xf);
    } else {
      pna_node_kernel<128, 128, 1664><<<N_NODES / 16, 256, 0, stream>>>(
          xb_in, xpost, wpostP, bpost, wlinP, blin, ln_g, ln_b, xb_out, xf);
    }
    unsigned short* tmp = xb_in; xb_in = xb_out; xb_out = tmp;
  }

  // global add pool
  zero_f32_k<<<(N_GRAPHS * 128 + 255) / 256, 256, 0, stream>>>(gpool, (long)N_GRAPHS * 128);
  pool_k<<<(int)(((long)N_NODES * 128 + 255) / 256), 256, 0, stream>>>(xf, batch, gpool);

  // head: mlp block (runs once) + output layers
  const float* mW1 = (const float*)d_in[34];
  const float* mW2 = (const float*)d_in[35];
  const float* mb1 = (const float*)d_in[36];
  const float* mb2 = (const float*)d_in[37];
  const float* mbe1 = (const float*)d_in[38];
  const float* mbe2 = (const float*)d_in[39];
  const float* mg1 = (const float*)d_in[40];
  const float* mg2 = (const float*)d_in[41];
  const float* oW1 = (const float*)d_in[42];
  const float* oW2 = (const float*)d_in[43];
  const float* oW3 = (const float*)d_in[44];
  const float* ob1 = (const float*)d_in[45];
  const float* ob2 = (const float*)d_in[46];
  const float* ob3 = (const float*)d_in[47];

  gemm_head_k<<<(N_GRAPHS * 128 + 255) / 256, 256, 0, stream>>>(gpool, mW1, mb1, h1,
                                                                N_GRAPHS, 128, 128, 0);
  ln_relu_head_k<<<(N_GRAPHS + 255) / 256, 256, 0, stream>>>(h1, mg1, mbe1, N_GRAPHS);
  gemm_head_k<<<(N_GRAPHS * 128 + 255) / 256, 256, 0, stream>>>(h1, mW2, mb2, h2,
                                                                N_GRAPHS, 128, 128, 0);
  ln_relu_head_k<<<(N_GRAPHS + 255) / 256, 256, 0, stream>>>(h2, mg2, mbe2, N_GRAPHS);
  gemm_head_k<<<(N_GRAPHS * 64 + 255) / 256, 256, 0, stream>>>(h2, oW1, ob1, o1,
                                                               N_GRAPHS, 128, 64, 1);
  gemm_head_k<<<(N_GRAPHS * 32 + 255) / 256, 256, 0, stream>>>(o1, oW2, ob2, o2,
                                                               N_GRAPHS, 64, 32, 1);
  gemm_head_k<<<(N_GRAPHS * 3 + 255) / 256, 256, 0, stream>>>(o2, oW3, ob3,
                                                              (float*)d_out,
                                                              N_GRAPHS, 32, 3, 2);
}
